// ProteoGNN_62526133895862
// MI455X (gfx1250) — matrix operand order repaired and checked by reading, not compiled, not run on hardware
//
#include <hip/hip_runtime.h>
#include <cstdint>

// ---------------------------------------------------------------------------
// ProteoGNN forward for MI455X (gfx1250, wave32).
// Dense GEMMs use v_wmma_f32_16x16x32_f16 (f16 inputs, f32 accumulate).
// B fragments are register-resident for the whole kernel; A streams via
// clause-grouped global_load_b128; C accesses are immediate-offset off one
// base pointer. Two row tiles per iteration -> two independent WMMA chains.
// Working set (~155MB) fits the 192MB L2.
// ---------------------------------------------------------------------------

typedef __attribute__((ext_vector_type(16))) _Float16 v16h;
typedef __attribute__((ext_vector_type(8)))  float    v8f;
typedef __attribute__((ext_vector_type(4)))  float    v4f;

#define BN_EPS 1e-5f

__device__ __forceinline__ void atomicAddF(float* p, float v) {
    unsafeAtomicAdd(p, v);   // native global_atomic_add_f32 on AMD
}

// ---------------------------------------------------------------------------
// Compute NT (1 or 2) vertically-adjacent 16x16 C tiles for one colTile.
//   Arow  = A + (rowTile0*16 + (lane&15)) * K          (per-lane row base)
//   Cbase = C + (rowTile0*16 + 8*(lane>>4)) * M + col  (per-lane store base)
// All addresses inside are compile-time immediate offsets from these bases.
// ---------------------------------------------------------------------------
template<int K, int M, bool ACC, bool RELU, int NT>
__device__ __forceinline__
void gemm_tiles(const float* __restrict__ Arow, float* __restrict__ Cbase,
                const v16h* breg, float bias, int half)
{
    v8f acc[NT];
#pragma unroll
    for (int tt = 0; tt < NT; ++tt)
#pragma unroll
        for (int i = 0; i < 8; ++i)
            acc[tt][i] = (ACC ? Cbase[(tt * 16 + i) * M] : 0.0f) + bias;

#pragma unroll
    for (int ks = 0; ks < K / 32; ++ks) {
#pragma unroll
        for (int tt = 0; tt < NT; ++tt) {
            // A fragment (16x32 f16): half=0 -> K {ks*32+0..7, +16..23},
            // half=1 -> K {ks*32+8..15, +24..31}; row = lane&15.
            const float* p0 = Arow + tt * 16 * K + ks * 32 + half * 8;
            v4f a00 = *(const v4f*)(p0);
            v4f a01 = *(const v4f*)(p0 + 4);
            v4f a10 = *(const v4f*)(p0 + 16);
            v4f a11 = *(const v4f*)(p0 + 20);
            v16h af;
#pragma unroll
            for (int j = 0; j < 4; ++j) {
                af[j]      = (_Float16)a00[j];
                af[4 + j]  = (_Float16)a01[j];
                af[8 + j]  = (_Float16)a10[j];
                af[12 + j] = (_Float16)a11[j];
            }
            acc[tt] = __builtin_amdgcn_wmma_f32_16x16x32_f16(
                          false, af, false, breg[ks], (short)0, acc[tt], false, false);
        }
    }

#pragma unroll
    for (int tt = 0; tt < NT; ++tt)
#pragma unroll
        for (int i = 0; i < 8; ++i) {
            float v = acc[tt][i];
            if (RELU) v = fmaxf(v, 0.0f);
            Cbase[(tt * 16 + i) * M] = v;
        }
}

// ---------------------------------------------------------------------------
// C[n x M] = (ACC ? C : 0) + bias[M] + A[n x K] @ W[K x M]   (optional ReLU)
// One wave owns one 16-col strip (colTile) and grid-strides over row-tile
// PAIRS. B fragments for all K-steps live in registers for the whole kernel.
// ---------------------------------------------------------------------------
template<int K, int M, bool ACC, bool RELU>
__global__ __launch_bounds__(256)
void gemm_wmma(const float* __restrict__ A, const float* __restrict__ W,
               const float* __restrict__ bias, float* __restrict__ C, int n)
{
    constexpr int KSTEPS = K / 32;
    constexpr int TILESM = M / 16;

    const int lane = threadIdx.x & 31;
    const int r    = lane & 15;
    const int half = lane >> 4;

    const int waveGlobal = blockIdx.x * 8 + (threadIdx.x >> 5);
    const int totalWaves = gridDim.x * 8;
    const int colTile    = waveGlobal % TILESM;     // waves 0..7 of a block share
    const int pairStart  = waveGlobal / TILESM;     // a row-tile stream (L1 reuse)
    const int pairStride = totalWaves / TILESM;     // integer for TILESM in {4,8}
    const int rowTiles   = (n + 15) >> 4;
    const int pairs      = (rowTiles + 1) >> 1;

    const int   col = colTile * 16 + r;
    const float b   = bias[col];

    // B fragments (32x16 f16 per k-step): lanes 0-15 hold K=ks*32+{0..7,16..23}
    // (col = lane&15), lanes 16-31 hold K=ks*32+{8..15,24..31}. Loaded ONCE.
    v16h breg[KSTEPS];
#pragma unroll
    for (int ks = 0; ks < KSTEPS; ++ks)
#pragma unroll
        for (int idx = 0; idx < 16; ++idx)
            breg[ks][idx] =
                (_Float16)W[(size_t)(ks * 32 + half * 8 + idx + (idx & 8)) * M + col];

    for (int pp = pairStart; pp < pairs; pp += pairStride) {
        const int rt0 = pp * 2;
        const float* __restrict__ Arow  = A + (size_t)(rt0 * 16 + r) * K;
        float*       __restrict__ Cbase = C + (size_t)(rt0 * 16 + 8 * half) * M + col;
        if (rt0 + 1 < rowTiles)
            gemm_tiles<K, M, ACC, RELU, 2>(Arow, Cbase, breg, b, half);
        else
            gemm_tiles<K, M, ACC, RELU, 1>(Arow, Cbase, breg, b, half);
    }
}

// ---------------------------------------------------------------------------
// BatchNorm stats: column-wise sum and sum-of-squares over N rows (H=128).
// ---------------------------------------------------------------------------
__global__ __launch_bounds__(128)
void bn_stats(const float* __restrict__ t, float* __restrict__ sums,
              float* __restrict__ sumsq, int n)
{
    const int col = threadIdx.x;   // 0..127
    float s = 0.0f, q = 0.0f;
    for (int row = blockIdx.x; row < n; row += gridDim.x) {
        float v = t[(size_t)row * 128 + col];
        s += v;
        q += v * v;
    }
    atomicAddF(&sums[col], s);
    atomicAddF(&sumsq[col], q);
}

// ---------------------------------------------------------------------------
// out = (RESID ? resid : 0) + relu( (t - mean) * gamma * rsqrt(var+eps) + beta )
// ---------------------------------------------------------------------------
template<bool RESID>
__global__ __launch_bounds__(256)
void bn_apply(const float* __restrict__ t, const float* __restrict__ sums,
              const float* __restrict__ sumsq, const float* __restrict__ gamma,
              const float* __restrict__ beta, const float* __restrict__ resid,
              float* __restrict__ out, int n)
{
    const size_t total  = (size_t)n * 128;
    const size_t stride = (size_t)gridDim.x * blockDim.x;
    const float  inv_n  = 1.0f / (float)n;
    for (size_t i = (size_t)blockIdx.x * blockDim.x + threadIdx.x; i < total; i += stride) {
        const int col = (int)(i & 127);
        const float mean = sums[col] * inv_n;
        const float var  = fmaxf(sumsq[col] * inv_n - mean * mean, 0.0f);
        float v = (t[i] - mean) * (gamma[col] * __frsqrt_rn(var + BN_EPS)) + beta[col];
        v = fmaxf(v, 0.0f);
        out[i] = RESID ? (resid[i] + v) : v;
    }
}

// ---------------------------------------------------------------------------
// Graph aggregation: nb[row[e]] += h[col[e]] ; deg[row[e]] += 1.  One wave/edge.
// ---------------------------------------------------------------------------
__global__ __launch_bounds__(256)
void scatter_edges(const long long* __restrict__ erow, const long long* __restrict__ ecol,
                   const float* __restrict__ h, float* __restrict__ nb,
                   float* __restrict__ deg, int E)
{
    const int lane   = threadIdx.x & 31;
    const int wave   = blockIdx.x * (blockDim.x >> 5) + (threadIdx.x >> 5);
    const int nwaves = gridDim.x * (blockDim.x >> 5);
    for (int e = wave; e < E; e += nwaves) {
        const long long dst = erow[e];
        const long long src = ecol[e];
        const float* __restrict__ s = h  + (size_t)src * 128;
        float*       __restrict__ d = nb + (size_t)dst * 128;
#pragma unroll
        for (int j = 0; j < 4; ++j)
            atomicAddF(&d[lane + 32 * j], s[lane + 32 * j]);
        if (lane == 0) atomicAddF(&deg[dst], 1.0f);
    }
}

__global__ __launch_bounds__(256)
void div_deg(float* __restrict__ nb, const float* __restrict__ deg, int n)
{
    const size_t total  = (size_t)n * 128;
    const size_t stride = (size_t)gridDim.x * blockDim.x;
    for (size_t i = (size_t)blockIdx.x * blockDim.x + threadIdx.x; i < total; i += stride)
        nb[i] *= (1.0f / fmaxf(deg[i >> 7], 1.0f));
}

__global__ __launch_bounds__(256)
void fill_zero(float* __restrict__ p, size_t n)
{
    const size_t stride = (size_t)gridDim.x * blockDim.x;
    for (size_t i = (size_t)blockIdx.x * blockDim.x + threadIdx.x; i < n; i += stride)
        p[i] = 0.0f;
}

// ---------------------------------------------------------------------------
// logits[i] = h2[i,:] . w + b   (h2 is N x 64)
// ---------------------------------------------------------------------------
__global__ __launch_bounds__(256)
void head_logits(const float* __restrict__ h2, const float* __restrict__ w,
                 const float* __restrict__ b, float* __restrict__ out, int n)
{
    const int i = blockIdx.x * blockDim.x + threadIdx.x;
    if (i >= n) return;
    const float* r = h2 + (size_t)i * 64;
    float s = b[0];
#pragma unroll
    for (int j = 0; j < 64; ++j) s = fmaf(r[j], w[j], s);
    out[i] = s;
}

// ---------------------------------------------------------------------------
// Host-side launch helper
// ---------------------------------------------------------------------------
template<int K, int M, bool ACC, bool RELU>
static inline void launch_gemm(const float* A, const float* W, const float* bias,
                               float* C, int n, hipStream_t stream)
{
    constexpr int TILESM = M / 16;
    const int rowTiles = (n + 15) / 16;
    const int pairs    = (rowTiles + 1) / 2;
    // ~4 pairs (8 row tiles) per wave; blocks*8 waves divisible by TILESM (4 or 8)
    int blocks = (pairs * TILESM + 31) / 32;
    if (blocks < 1) blocks = 1;
    gemm_wmma<K, M, ACC, RELU><<<blocks, 256, 0, stream>>>(A, W, bias, C, n);
}

extern "C" void kernel_launch(void* const* d_in, const int* in_sizes, int n_in,
                              void* d_out, int out_size, void* d_ws, size_t ws_size,
                              hipStream_t stream)
{
    const int N = in_sizes[0] / 64;       // 100000
    const int E = in_sizes[1] / 2;        // 1600000
    const int H = 128;

    const float*     x    = (const float*)d_in[0];
    const long long* erow = (const long long*)d_in[1];     // edge_index is int64
    const long long* ecol = erow + E;

    // params flattened in setup_inputs() insertion order
    int p = 2;
    const float* enc_w1  = (const float*)d_in[p++];
    const float* enc_b1  = (const float*)d_in[p++];
    const float* enc_g1  = (const float*)d_in[p++];
    const float* enc_be1 = (const float*)d_in[p++];
    const float* enc_w2  = (const float*)d_in[p++];
    const float* enc_b2  = (const float*)d_in[p++];
    const float* proj_w  = (const float*)d_in[p++];
    const float* proj_b  = (const float*)d_in[p++];
    const float *lws[4], *lbs[4], *lwn[4], *lbn[4], *lg[4], *lbe[4];
    for (int l = 0; l < 4; ++l) {
        lws[l] = (const float*)d_in[p++];
        lbs[l] = (const float*)d_in[p++];
        lwn[l] = (const float*)d_in[p++];
        lbn[l] = (const float*)d_in[p++];
        lg[l]  = (const float*)d_in[p++];
        lbe[l] = (const float*)d_in[p++];
    }
    const float* head_w1 = (const float*)d_in[p++];
    const float* head_b1 = (const float*)d_in[p++];
    const float* head_w2 = (const float*)d_in[p++];
    const float* head_b2 = (const float*)d_in[p++];

    // Workspace carve (floats): h, t, nb are N*128 each; then deg (N), sums+sumsq.
    const size_t NH = (size_t)N * H;
    float* h     = (float*)d_ws;
    float* t     = h + NH;
    float* nb    = t + NH;
    float* deg   = nb + NH;
    float* sums  = deg + N;
    float* sumsq = sums + H;

    float* logits = (float*)d_out;

    const int FILL_BLK = 4096, STAT_BLK = 1024, APPLY_BLK = 8192, SCAT_BLK = 8192;

    // ---- NodeEncoder: h = relu(BN(x @ enc_w1 + b1)) ----
    launch_gemm<64, 128, false, false>(x, enc_w1, enc_b1, t, N, stream);
    fill_zero<<<1, 256, 0, stream>>>(sums, 2 * H);
    bn_stats<<<STAT_BLK, 128, 0, stream>>>(t, sums, sumsq, N);
    bn_apply<false><<<APPLY_BLK, 256, 0, stream>>>(t, sums, sumsq, enc_g1, enc_be1,
                                                   nullptr, h, N);
    // h = h @ enc_w2 + b2 ; h = h @ proj_w + proj_b
    launch_gemm<128, 128, false, false>(h, enc_w2, enc_b2, t, N, stream);
    launch_gemm<128, 128, false, false>(t, proj_w, proj_b, h, N, stream);

    // ---- Residual GNN layers ----
    for (int l = 0; l < 4; ++l) {
        // t = h @ ws + bs
        launch_gemm<128, 128, false, false>(h, lws[l], lbs[l], t, N, stream);
        // nb_mean
        fill_zero<<<FILL_BLK, 256, 0, stream>>>(nb, NH);
        fill_zero<<<FILL_BLK, 256, 0, stream>>>(deg, (size_t)N);
        scatter_edges<<<SCAT_BLK, 256, 0, stream>>>(erow, ecol, h, nb, deg, E);
        div_deg<<<APPLY_BLK, 256, 0, stream>>>(nb, deg, N);
        // t += nb @ wn + bn
        launch_gemm<128, 128, true, false>(nb, lwn[l], lbn[l], t, N, stream);
        // h = h + relu(BN(t))
        fill_zero<<<1, 256, 0, stream>>>(sums, 2 * H);
        bn_stats<<<STAT_BLK, 128, 0, stream>>>(t, sums, sumsq, N);
        bn_apply<true><<<APPLY_BLK, 256, 0, stream>>>(t, sums, sumsq, lg[l], lbe[l],
                                                      h, h, N);
    }

    // ---- Head: h2 = relu(h @ head_w1 + b1) ; logits = h2 @ head_w2 + b2 ----
    launch_gemm<128, 64, false, true>(h, head_w1, head_b1,
                                      nb /*reuse as h2 (N x 64)*/, N, stream);
    head_logits<<<(N + 255) / 256, 256, 0, stream>>>(nb, head_w2, head_b2, logits, N);
}
